// LSTMAutoencoder_48086453846063
// MI455X (gfx1250) — matrix operand order repaired and checked by reading, not compiled, not run on hardware
//
#include <hip/hip_runtime.h>
#include <hip/hip_bf16.h>

typedef __attribute__((ext_vector_type(16))) _Float16 v16h;
typedef __attribute__((ext_vector_type(8)))  _Float16 v8h;
typedef __attribute__((ext_vector_type(8)))  float    v8f;

#define B_N 4096
#define T_N 128
#define D_N 21
#define H_N 64
#define L_N 32

// ---------------------------------------------------------------- helpers

__device__ __forceinline__ float sigm(float x) {
    x = fminf(fmaxf(x, -30.f), 30.f);
    return 1.f / (1.f + __expf(-x));
}
__device__ __forceinline__ float tanh_f(float x) {
    x = fminf(fmaxf(x, -15.f), 15.f);
    float e = __expf(2.f * x);
    return (e - 1.f) / (e + 1.f);
}
__device__ __forceinline__ v8f splat8(float x) {
    v8f r;
#pragma unroll
    for (int e = 0; e < 8; ++e) r[e] = x;
    return r;
}

// A fragment (16x32 f16, one 16-row tile, K window kb..kb+31).
// Lane<16: elems 0..7 = K kb..kb+7, elems 8..15 = K kb+16..kb+23 of row (lane&15).
// Lane>=16: same +8 in K.  Two contiguous 16B chunks per lane (2x b128).
__device__ __forceinline__ v16h load_afrag(const _Float16* __restrict__ base,
                                           int rowStride, int kb, int lane) {
    const _Float16* rp = base + (lane & 15) * rowStride + kb + ((lane >> 4) << 3);
    v8h lo = *(const v8h*)(rp);
    v8h hi = *(const v8h*)(rp + 16);
    v16h r;
#pragma unroll
    for (int e = 0; e < 8; ++e) { r[e] = lo[e]; r[e + 8] = hi[e]; }
    return r;
}

// B fragment (32x16 f16) pre-packed so each lane reads 16 contiguous halves (32B).
__device__ __forceinline__ v16h load_bfrag(const _Float16* __restrict__ packed,
                                           int fragIdx, int lane) {
    const _Float16* p = packed + ((size_t)fragIdx << 9) + (lane << 4);
    return *(const v16h*)p;
}

__device__ __forceinline__ v8f wmma_f16(v16h a, v16h b, v8f c) {
    return __builtin_amdgcn_wmma_f32_16x16x32_f16(false, a, false, b,
                                                  (short)0, c, false, false);
}

// ---------------------------------------------------------------- packing

// Pack W[Nvalid x Kvalid] (row-major, leading dim Kld) as B=W^T fragments:
// frag (ntile*KT + ktile): lane<16 -> col ntile*16+lane, K ktile*32+{0..7,16..23}
//                          lane>=16 -> col ntile*16+(lane-16), K ktile*32+{8..15,24..31}
__global__ void pack_bfrag(const float* __restrict__ W, _Float16* __restrict__ dst,
                           int Nvalid, int Kvalid, int Kld, int NT, int KT) {
    int idx = blockIdx.x * blockDim.x + threadIdx.x;
    int total = NT * KT * 512;
    if (idx >= total) return;
    int frag   = idx >> 9;
    int within = idx & 511;
    int lane   = within >> 4;
    int i      = within & 15;
    int ntile  = frag / KT;
    int ktile  = frag - ntile * KT;
    int c = lane & 15;
    int k = ktile * 32 + ((lane >> 4) << 3) + (i & 7) + ((i >> 3) << 4);
    int n = ntile * 16 + c;
    float v = 0.f;
    if (n < Nvalid && k < Kvalid) v = W[(size_t)n * Kld + k];
    dst[idx] = (_Float16)v;
}

__global__ void add_vec(const float* __restrict__ a, const float* __restrict__ b,
                        float* __restrict__ dst, int n) {
    int i = blockIdx.x * blockDim.x + threadIdx.x;
    if (i < n) dst[i] = a[i] + b[i];
}

__global__ void copy_pad(const float* __restrict__ a, float* __restrict__ dst,
                         int nvalid, int ntot) {
    int i = blockIdx.x * blockDim.x + threadIdx.x;
    if (i < ntot) dst[i] = (i < nvalid) ? a[i] : 0.f;
}

// x [B,T,21] f32 -> xh [B,T,32] f16 zero-padded in K
__global__ void convert_x(const float* __restrict__ x, _Float16* __restrict__ xh) {
    int idx = blockIdx.x * blockDim.x + threadIdx.x;
    const int total = B_N * T_N * 32;
    if (idx >= total) return;
    int k = idx & 31;
    size_t bt = (size_t)(idx >> 5);
    xh[idx] = (k < D_N) ? (_Float16)x[bt * D_N + k] : (_Float16)0.f;
}

// ---------------------------------------------------------------- LSTM layer
// MODE 0: encoder L0  (zero init, store full seq f16)
// MODE 1: encoder L1  (zero init, store last-step h only)
// MODE 2: decoder L0  (h0 init, read x time-reversed, store full seq f16)
// MODE 3: decoder L1  (h0 init, fused output projection -> rec f32, time re-reversed)
template <int KIN, int MODE>
__global__ __launch_bounds__(128) void lstm_layer(
    const _Float16* __restrict__ xin,   // [B,T,KIN] f16
    const _Float16* __restrict__ wihp,  // packed B frags, NT=16, KT=KIN/32
    const _Float16* __restrict__ whhp,  // packed B frags, NT=16, KT=2
    const float*    __restrict__ bias,  // [256] (bih+bhh)
    const _Float16* __restrict__ h0g,   // [B,64] f16 or null
    _Float16*       __restrict__ oseq,  // MODE0/2: [B,T,64]; MODE1: [B,64]
    const _Float16* __restrict__ outwp, // MODE3: packed NT=2,KT=2
    const float*    __restrict__ outbp, // MODE3: [32]
    float*          __restrict__ rec)   // MODE3: [B,T,21]
{
    constexpr int KT = KIN / 32;
    __shared__ __align__(16) _Float16 hbuf[2][16][72];       // padded stride
    __shared__ __align__(16) _Float16 wihs[16 * KT * 512];

    const int tid   = threadIdx.x;
    const int j     = tid >> 5;        // wave id 0..3
    const int lane  = tid & 31;
    const int col16 = lane & 15;
    const int hiH   = lane >> 4;
    const int bbase = blockIdx.x * 16;

    // stage Wih fragments into LDS
    {
        const int n16 = (16 * KT * 512) / 8;
        const uint4* s = (const uint4*)wihp;
        uint4* d = (uint4*)wihs;
        for (int i = tid; i < n16; i += 128) d[i] = s[i];
    }
    // init h (zero or h0)
    for (int i = tid; i < 16 * 64; i += 128) {
        int r = i >> 6, c = i & 63;
        _Float16 v = (_Float16)0.f;
        if (MODE >= 2) v = h0g[(size_t)(bbase + r) * 64 + c];
        hbuf[0][r][c] = v;
    }
    __syncthreads();

    // Whh fragments resident in VGPRs for the entire time loop
    v16h whf[4][2];
#pragma unroll
    for (int gi = 0; gi < 4; ++gi)
#pragma unroll
        for (int kt = 0; kt < 2; ++kt)
            whf[gi][kt] = load_bfrag(whhp, (4 * gi + j) * 2 + kt, lane);

    float bgate[4];
#pragma unroll
    for (int gi = 0; gi < 4; ++gi) bgate[gi] = bias[gi * 64 + j * 16 + col16];

    float bo_ = 0.f;
    v16h owf0, owf1;
    if (MODE == 3 && j < 2) {
        bo_  = outbp[j * 16 + col16];
        owf0 = load_bfrag(outwp, j * 2 + 0, lane);
        owf1 = load_bfrag(outwp, j * 2 + 1, lane);
    }

    v8f creg = splat8(0.f);

    for (int t = 0; t < T_N; ++t) {
        const int rb = t & 1, wb = rb ^ 1;
        const int tmem = (MODE == 2) ? (T_N - 1 - t) : t;

        // input A fragments straight from global (L2-resident sequences)
        const _Float16* xb = xin + ((size_t)bbase * T_N + tmem) * KIN;
        v16h ax[KT];
#pragma unroll
        for (int kt = 0; kt < KT; ++kt)
            ax[kt] = load_afrag(xb, T_N * KIN, kt * 32, lane);

        // recurrent A fragments from LDS
        v16h ah[2];
#pragma unroll
        for (int kt = 0; kt < 2; ++kt)
            ah[kt] = load_afrag(&hbuf[rb][0][0], 72, kt * 32, lane);

        v8f acc[4];
#pragma unroll
        for (int gi = 0; gi < 4; ++gi) {
            acc[gi] = splat8(bgate[gi]);
#pragma unroll
            for (int kt = 0; kt < KT; ++kt)
                acc[gi] = wmma_f16(ax[kt],
                                   load_bfrag(wihs, (4 * gi + j) * KT + kt, lane),
                                   acc[gi]);
#pragma unroll
            for (int kt = 0; kt < 2; ++kt)
                acc[gi] = wmma_f16(ah[kt], whf[gi][kt], acc[gi]);
        }

        // lane-local cell update: this wave owns i,f,g,o for its 16 h-units
        float hv[8];
#pragma unroll
        for (int e = 0; e < 8; ++e) {
            float iv = sigm(acc[0][e]);
            float fv = sigm(acc[1][e]);
            float gv = tanh_f(acc[2][e]);
            float ov = sigm(acc[3][e]);
            float cv = fv * creg[e] + iv * gv;
            creg[e] = cv;
            hv[e] = ov * tanh_f(cv);
        }

#pragma unroll
        for (int e = 0; e < 8; ++e) {
            int row = e + 8 * hiH;
            hbuf[wb][row][16 * j + col16] = (_Float16)hv[e];
        }
        if (MODE == 0 || MODE == 2) {
#pragma unroll
            for (int e = 0; e < 8; ++e) {
                int row = e + 8 * hiH;
                oseq[((size_t)(bbase + row) * T_N + t) * 64 + 16 * j + col16] =
                    (_Float16)hv[e];
            }
        } else if (MODE == 1) {
            if (t == T_N - 1) {
#pragma unroll
                for (int e = 0; e < 8; ++e) {
                    int row = e + 8 * hiH;
                    oseq[(size_t)(bbase + row) * 64 + 16 * j + col16] =
                        (_Float16)hv[e];
                }
            }
        }
        __syncthreads();

        if (MODE == 3) {
            if (j < 2) {  // wave-uniform: EXEC stays all-ones for WMMA
                v16h a0 = load_afrag(&hbuf[wb][0][0], 72, 0, lane);
                v16h a1 = load_afrag(&hbuf[wb][0][0], 72, 32, lane);
                v8f oacc = splat8(bo_);
                oacc = wmma_f16(a0, owf0, oacc);
                oacc = wmma_f16(a1, owf1, oacc);
                int colg = 16 * j + col16;
                if (colg < D_N) {
#pragma unroll
                    for (int e = 0; e < 8; ++e) {
                        int row = e + 8 * hiH;
                        rec[((size_t)(bbase + row) * T_N + (T_N - 1 - t)) * D_N +
                            colg] = oacc[e];
                    }
                }
            }
        }
    }
}

// ---------------------------------------------------------------- latent
// latent = hT2 @ tlW^T + tlb (f32 -> d_out tail); h0 = latent @ flW^T + flb (f16)
__global__ __launch_bounds__(128) void latent_kernel(
    const _Float16* __restrict__ hT2,
    const _Float16* __restrict__ tlWp, const float* __restrict__ tlbp,
    const _Float16* __restrict__ flWp, const float* __restrict__ flbp,
    float* __restrict__ lat_out, _Float16* __restrict__ h0out)
{
    __shared__ __align__(16) _Float16 latf[4][16][40];
    const int tid = threadIdx.x, j = tid >> 5, lane = tid & 31;
    const int col16 = lane & 15, hiH = lane >> 4;
    const int bbase = blockIdx.x * 64 + j * 16;

    v16h a0 = load_afrag(hT2 + (size_t)bbase * 64, 64, 0, lane);
    v16h a1 = load_afrag(hT2 + (size_t)bbase * 64, 64, 32, lane);
#pragma unroll
    for (int nt = 0; nt < 2; ++nt) {
        v8f acc = splat8(tlbp[nt * 16 + col16]);
        acc = wmma_f16(a0, load_bfrag(tlWp, nt * 2 + 0, lane), acc);
        acc = wmma_f16(a1, load_bfrag(tlWp, nt * 2 + 1, lane), acc);
#pragma unroll
        for (int e = 0; e < 8; ++e) {
            int row = e + 8 * hiH;
            lat_out[(size_t)(bbase + row) * L_N + nt * 16 + col16] = acc[e];
            latf[j][row][nt * 16 + col16] = (_Float16)acc[e];
        }
    }
    __syncthreads();
    v16h al = load_afrag(&latf[j][0][0], 40, 0, lane);
#pragma unroll
    for (int nt = 0; nt < 4; ++nt) {
        v8f acc = splat8(flbp[nt * 16 + col16]);
        acc = wmma_f16(al, load_bfrag(flWp, nt, lane), acc);
#pragma unroll
        for (int e = 0; e < 8; ++e) {
            int row = e + 8 * hiH;
            h0out[(size_t)(bbase + row) * 64 + nt * 16 + col16] = (_Float16)acc[e];
        }
    }
}

// ---------------------------------------------------------------- host

extern "C" void kernel_launch(void* const* d_in, const int* in_sizes, int n_in,
                              void* d_out, int out_size, void* d_ws, size_t ws_size,
                              hipStream_t stream) {
    (void)in_sizes; (void)n_in; (void)out_size; (void)ws_size;
    const float* x     = (const float*)d_in[0];
    const float* eWih0 = (const float*)d_in[1];
    const float* eWhh0 = (const float*)d_in[2];
    const float* ebih0 = (const float*)d_in[3];
    const float* ebhh0 = (const float*)d_in[4];
    const float* eWih1 = (const float*)d_in[5];
    const float* eWhh1 = (const float*)d_in[6];
    const float* ebih1 = (const float*)d_in[7];
    const float* ebhh1 = (const float*)d_in[8];
    const float* dWih0 = (const float*)d_in[9];
    const float* dWhh0 = (const float*)d_in[10];
    const float* dbih0 = (const float*)d_in[11];
    const float* dbhh0 = (const float*)d_in[12];
    const float* dWih1 = (const float*)d_in[13];
    const float* dWhh1 = (const float*)d_in[14];
    const float* dbih1 = (const float*)d_in[15];
    const float* dbhh1 = (const float*)d_in[16];
    const float* tlW   = (const float*)d_in[17];
    const float* tlb   = (const float*)d_in[18];
    const float* flW   = (const float*)d_in[19];
    const float* flb   = (const float*)d_in[20];
    const float* outW  = (const float*)d_in[21];
    const float* outb  = (const float*)d_in[22];

    char* base = (char*)d_ws;
    size_t off = 0;
    auto take = [&](size_t bytes) -> char* {
        char* p = base + off;
        off = (off + bytes + 255) & ~(size_t)255;
        return p;
    };

    _Float16* xh    = (_Float16*)take((size_t)B_N * T_N * 32 * 2);
    _Float16* seq   = (_Float16*)take((size_t)B_N * T_N * 64 * 2);
    _Float16* hT2   = (_Float16*)take((size_t)B_N * 64 * 2);
    _Float16* h0b   = (_Float16*)take((size_t)B_N * 64 * 2);
    _Float16* wih0p = (_Float16*)take((size_t)16 * 1 * 512 * 2);
    _Float16* whh0p = (_Float16*)take((size_t)16 * 2 * 512 * 2);
    _Float16* wih1p = (_Float16*)take((size_t)16 * 2 * 512 * 2);
    _Float16* whh1p = (_Float16*)take((size_t)16 * 2 * 512 * 2);
    _Float16* wih2p = (_Float16*)take((size_t)16 * 1 * 512 * 2);
    _Float16* whh2p = (_Float16*)take((size_t)16 * 2 * 512 * 2);
    _Float16* wih3p = (_Float16*)take((size_t)16 * 2 * 512 * 2);
    _Float16* whh3p = (_Float16*)take((size_t)16 * 2 * 512 * 2);
    _Float16* tlWp  = (_Float16*)take((size_t)2 * 2 * 512 * 2);
    _Float16* flWp  = (_Float16*)take((size_t)4 * 1 * 512 * 2);
    _Float16* outWp = (_Float16*)take((size_t)2 * 2 * 512 * 2);
    float* bs0   = (float*)take(256 * 4);
    float* bs1   = (float*)take(256 * 4);
    float* bs2   = (float*)take(256 * 4);
    float* bs3   = (float*)take(256 * 4);
    float* tlbp  = (float*)take(32 * 4);
    float* flbp  = (float*)take(64 * 4);
    float* outbp = (float*)take(32 * 4);

    auto gpk = [](int tot) { return dim3((tot + 255) / 256); };
    pack_bfrag<<<gpk(16 * 1 * 512), 256, 0, stream>>>(eWih0, wih0p, 256, 21, 21, 16, 1);
    pack_bfrag<<<gpk(16 * 2 * 512), 256, 0, stream>>>(eWhh0, whh0p, 256, 64, 64, 16, 2);
    pack_bfrag<<<gpk(16 * 2 * 512), 256, 0, stream>>>(eWih1, wih1p, 256, 64, 64, 16, 2);
    pack_bfrag<<<gpk(16 * 2 * 512), 256, 0, stream>>>(eWhh1, whh1p, 256, 64, 64, 16, 2);
    pack_bfrag<<<gpk(16 * 1 * 512), 256, 0, stream>>>(dWih0, wih2p, 256, 21, 21, 16, 1);
    pack_bfrag<<<gpk(16 * 2 * 512), 256, 0, stream>>>(dWhh0, whh2p, 256, 64, 64, 16, 2);
    pack_bfrag<<<gpk(16 * 2 * 512), 256, 0, stream>>>(dWih1, wih3p, 256, 64, 64, 16, 2);
    pack_bfrag<<<gpk(16 * 2 * 512), 256, 0, stream>>>(dWhh1, whh3p, 256, 64, 64, 16, 2);
    pack_bfrag<<<gpk(2 * 2 * 512), 256, 0, stream>>>(tlW, tlWp, 32, 64, 64, 2, 2);
    pack_bfrag<<<gpk(4 * 1 * 512), 256, 0, stream>>>(flW, flWp, 64, 32, 32, 4, 1);
    pack_bfrag<<<gpk(2 * 2 * 512), 256, 0, stream>>>(outW, outWp, 21, 64, 64, 2, 2);

    add_vec<<<1, 256, 0, stream>>>(ebih0, ebhh0, bs0, 256);
    add_vec<<<1, 256, 0, stream>>>(ebih1, ebhh1, bs1, 256);
    add_vec<<<1, 256, 0, stream>>>(dbih0, dbhh0, bs2, 256);
    add_vec<<<1, 256, 0, stream>>>(dbih1, dbhh1, bs3, 256);
    copy_pad<<<1, 64, 0, stream>>>(tlb, tlbp, 32, 32);
    copy_pad<<<1, 64, 0, stream>>>(flb, flbp, 64, 64);
    copy_pad<<<1, 64, 0, stream>>>(outb, outbp, 21, 32);

    convert_x<<<gpk(B_N * T_N * 32), 256, 0, stream>>>(x, xh);

    float* out = (float*)d_out;
    const dim3 lgrid(B_N / 16), lblk(128);

    // encoder L0: x(21->pad32) -> h1 sequence
    lstm_layer<32, 0><<<lgrid, lblk, 0, stream>>>(xh, wih0p, whh0p, bs0, nullptr,
                                                  seq, nullptr, nullptr, nullptr);
    // encoder L1: h1 -> hT2 only
    lstm_layer<64, 1><<<lgrid, lblk, 0, stream>>>(seq, wih1p, whh1p, bs1, nullptr,
                                                  hT2, nullptr, nullptr, nullptr);
    // latent + decoder init h0
    latent_kernel<<<B_N / 64, 128, 0, stream>>>(hT2, tlWp, tlbp, flWp, flbp,
                                                out + (size_t)B_N * T_N * D_N, h0b);
    // decoder L0: reversed x, h0 init -> d1 sequence (overwrites h1 buffer)
    lstm_layer<32, 2><<<lgrid, lblk, 0, stream>>>(xh, wih2p, whh2p, bs2, h0b,
                                                  seq, nullptr, nullptr, nullptr);
    // decoder L1: d1, h0 init, fused projection -> rec (time re-reversed)
    lstm_layer<64, 3><<<lgrid, lblk, 0, stream>>>(seq, wih3p, whh3p, bs3, h0b,
                                                  nullptr, outWp, outbp, out);
}
